// ActorCriticNetwork_64828236366284
// MI455X (gfx1250) — matrix operand order repaired and checked by reading, not compiled, not run on hardware
//
#include <hip/hip_runtime.h>
#include <hip/hip_bf16.h>

// ---------------------------------------------------------------------------
// ActorCritic: LSTM(1024, sigmoid) -> LSTM(512, sigmoid) -> FC head
// B=32, T=1024, FEAT=64.  f16 WMMA (v_wmma_f32_16x16x32_f16), f32 accum.
// Persistent scan kernels: recurrent weights resident in LDS (loaded via
// global_load_async_to_lds_b128), per-step h broadcast via TDM
// tensor_load_to_lds, grid barrier per timestep.
// ---------------------------------------------------------------------------

typedef __attribute__((ext_vector_type(16))) _Float16 v16h;
typedef __attribute__((ext_vector_type(8)))  float    v8f;
typedef __attribute__((ext_vector_type(4)))  unsigned v4u;
typedef __attribute__((ext_vector_type(8)))  int      v8i;
typedef __attribute__((ext_vector_type(4)))  int      v4i;

#define NWG1 64
#define NWG2 32

// ---- workspace layout (bytes) ---------------------------------------------
#define OFF_BAR1   0            // 1024 ints            (4 KB)
#define OFF_BAR2   4096         // 1024 ints            (4 KB)
#define OFF_HB1    8192         // 2 * 32768 halves     (128 KB)  h1 ping-pong (A-frag)
#define OFF_HB2    139264       // 2 * 16384 halves     (64 KB)   h2 ping-pong (A-frag)
#define ZERO_WORDS 51200        // [0, 204800) zeroed every launch
#define OFF_H2F    204800       // 32*512 f32           (64 KB)   final h2
#define OFF_PX     270336       // 1024*2048 halves     (4 MB)    x A-frags per t
#define OFF_PW1K   4464640      // 256*2*512 halves     (0.5 MB)
#define OFF_PW1R   4988928      // 256*32*512 halves    (8 MB)
#define OFF_PW2K   13377536     // 128*32*512 halves    (4 MB)
#define OFF_PW2R   17571840     // 128*16*512 halves    (2 MB)
#define OFF_SEQ    19668992     // 1024*32768 halves    (64 MB)   h1 sequence (A-frag)

// ---------------------------------------------------------------------------
// gfx1250 data-movement helpers
// ---------------------------------------------------------------------------
__device__ __forceinline__ unsigned lds_off_of(const void* p) {
  // generic LDS pointer = {shared aperture hi, LDS byte offset lo}
  return (unsigned)(unsigned long long)p;
}

// Async global -> LDS, 16 B per active lane, tracked by ASYNCcnt.
__device__ __forceinline__ void async_ld_b128(void* lds, const void* g) {
  asm volatile("global_load_async_to_lds_b128 %0, %1, off"
               :
               : "v"(lds_off_of(lds)), "v"(g)
               : "memory");
}
__device__ __forceinline__ void s_wait_async0() {
  asm volatile("s_wait_asynccnt 0x0" ::: "memory");
}

// TDM: 1-D contiguous tile load global -> LDS (data_size = 8 B elements).
// One instruction per wave; tracked by TENSORcnt; EXEC ignored.
__device__ __forceinline__ void tdm_load_1d(void* lds, const void* g,
                                            unsigned bytes) {
  unsigned long long ga = (unsigned long long)g;
  unsigned n8 = bytes >> 3;  // 8-byte elements, fits tile_dim0 (16 bits)
  v4u g0;
  g0.x = 1u;                                   // count=1, user descriptor
  g0.y = lds_off_of(lds);                      // lds_addr [63:32]
  g0.z = (unsigned)(ga & 0xffffffffu);         // global_addr [95:64]
  g0.w = (unsigned)((ga >> 32) & 0x01ffffffu)  // global_addr [120:96]
         | (2u << 30);                         // type = 2 ("image")
  v8i g1;
  g1[0] = (int)(3u << 16);                     // wg_mask=0, data_size=3 (8B)
  g1[1] = (int)((n8 & 0xffffu) << 16);         // tensor_dim0 [15:0]
  g1[2] = (int)((n8 >> 16) | (1u << 16));      // tensor_dim0 [31:16], dim1=1
  g1[3] = (int)((n8 & 0xffffu) << 16);         // tile_dim0 (bits 127:112)
  g1[4] = 0;                                   // tile_dim1/2 unused
  g1[5] = (int)n8;                             // tensor_dim0_stride [31:0]
  g1[6] = 0;
  g1[7] = 0;
  v4i z = {0, 0, 0, 0};
#if defined(__clang_major__) && (__clang_major__ >= 23)
  v8i z8 = {0, 0, 0, 0, 0, 0, 0, 0};
  __builtin_amdgcn_tensor_load_to_lds(g0, g1, z, z, z8, 0);
#else
  __builtin_amdgcn_tensor_load_to_lds(g0, g1, z, z, 0);
#endif
}

// ---------------------------------------------------------------------------
__global__ __launch_bounds__(256) void zero_ws_kernel(unsigned int* w) {
  unsigned int i = blockIdx.x * 256u + threadIdx.x;
  if (i < ZERO_WORDS) w[i] = 0u;
}

// Pack f32 row-major W[K,N] into f16 WMMA B fragments:
//   frag = ntile*KT + ktile ; element = lane*16 + j
//   lanes 0-15: K = ktile*32 + (j<8 ? j : 8+j),   n = ntile*16 + lane
//   lanes16-31: K = ktile*32 + (j<8 ? 8+j : 16+j), n = ntile*16 + (lane-16)
__global__ __launch_bounds__(256) void pack_b_kernel(const float* __restrict__ src,
                                                     _Float16* __restrict__ dst,
                                                     int K, int N) {
  int tid = blockIdx.x * 256 + threadIdx.x;
  if (tid >= K * N) return;
  int KT = K >> 5;
  int fragElems = KT << 9;
  int ntile = tid / fragElems;
  int rem   = tid - ntile * fragElems;
  int ktile = rem >> 9;
  int e     = rem & 511;
  int lane  = e >> 4, j = e & 15;
  int n = (ntile << 4) + (lane & 15);
  int k = (ktile << 5) + ((j >> 3) << 4) + ((lane >> 4) << 3) + (j & 7);
  dst[tid] = (_Float16)src[k * N + n];
}

// Pack state [B=32][T=1024][F=64] into per-t A fragments [T][2 mt][2 kt][512].
__global__ __launch_bounds__(256) void pack_x_kernel(const float* __restrict__ src,
                                                     _Float16* __restrict__ dst) {
  int tid = blockIdx.x * 256 + threadIdx.x;  // < 32*1024*64
  if (tid >= 32 * 1024 * 64) return;
  int t = tid >> 11;
  int r = tid & 2047;
  int m = r >> 6;
  int k = r & 63;
  int mtile = m >> 4, mr = m & 15;
  int ktile = k >> 5, kl = k & 31;
  int half = (kl >> 3) & 1;
  int j    = (kl & 7) + (((kl >> 4) & 1) << 3);
  int lane = mr + (half << 4);
  dst[(t << 11) + ((((mtile << 1) + ktile)) << 9) + (lane << 4) + j] =
      (_Float16)src[(m * 1024 + t) * 64 + k];
}

__device__ __forceinline__ float sigf(float x) {
  return 1.0f / (1.0f + __expf(-x));
}

__device__ __forceinline__ void grid_barrier(int* bar, int t, int nwg) {
  __threadfence();
  __syncthreads();
  if (threadIdx.x == 0) {
    __hip_atomic_fetch_add(bar + t, 1, __ATOMIC_RELEASE, __HIP_MEMORY_SCOPE_AGENT);
    while (__hip_atomic_load(bar + t, __ATOMIC_ACQUIRE, __HIP_MEMORY_SCOPE_AGENT) < nwg)
      __builtin_amdgcn_s_sleep(2);
  }
  __syncthreads();
}

// A-fragment index for element (row m, column kcol) with KT ktiles of 32.
__device__ __forceinline__ int a_frag_idx(int m, int kcol, int KT) {
  int mtile = m >> 4, mr = m & 15;
  int ktile = kcol >> 5, kl = kcol & 31;
  int half = (kl >> 3) & 1;
  int j    = (kl & 7) + (((kl >> 4) & 1) << 3);
  int lane = mr + (half << 4);
  return ((mtile * KT + ktile) << 9) + (lane << 4) + j;
}

// ---------------------------------------------------------------------------
// LSTM1 persistent scan.  64 WGs x 256 threads.  WG owns gate-cols [wg*16,+16)
// of each of the 4 gates.  LDS: W1r shard (128K) + W1k shard (8K) + h A-frags
// (64K) + z exchange (8K) + c state (2K) = 210 KB.
// ---------------------------------------------------------------------------
__global__ __launch_bounds__(256) void lstm1_kernel(
    const _Float16* __restrict__ px,   // [1024][2048]
    const _Float16* __restrict__ pWk,  // [256][2][512]
    const _Float16* __restrict__ pWr,  // [256][32][512]
    const float*    __restrict__ b1,   // [4096]
    _Float16*       __restrict__ hbuf, // [2][32768]
    _Float16*       __restrict__ seq,  // [1024][32768]
    int*            __restrict__ bar) {
  extern __shared__ __align__(16) char smem[];
  _Float16* sW  = (_Float16*)smem;           // 4*32*512
  _Float16* sWk = sW  + 4 * 32 * 512;        // 4*2*512
  _Float16* sA  = sWk + 4 * 2 * 512;         // 2*32*512
  float*    sZ  = (float*)(sA + 2 * 32 * 512);  // 8*256
  float*    sC  = sZ + 8 * 256;              // 512

  const int tid  = threadIdx.x;
  const int wg   = blockIdx.x;
  const int wave = tid >> 5;
  const int lane = tid & 31;
  const int gi   = wave >> 1;  // gate 0..3
  const int mi   = wave & 1;   // M-tile 0..1

  // async-load resident weight shards (f16 fragments) into LDS
  for (int g = 0; g < 4; ++g) {
    const char* s = (const char*)(pWr + (((g * 64 + wg) * 32) << 9));
    char* d = (char*)(sW + ((g * 32) << 9));
    for (int i = tid * 16; i < 32 * 512 * 2; i += 256 * 16)
      async_ld_b128(d + i, s + i);
  }
  for (int g = 0; g < 4; ++g) {
    const char* s = (const char*)(pWk + (((g * 64 + wg) * 2) << 9));
    char* d = (char*)(sWk + ((g * 2) << 9));
    for (int i = tid * 16; i < 2 * 512 * 2; i += 256 * 16)
      async_ld_b128(d + i, s + i);
  }
  sC[tid] = 0.0f;
  sC[tid + 256] = 0.0f;
  s_wait_async0();
  __syncthreads();

  for (int t = 0; t < 1024; ++t) {
    // broadcast h_{t-1} fragments into LDS via TDM (one DMA per step)
    if (wave == 0) {
      tdm_load_1d(sA, hbuf + ((t & 1) * 32768), 65536);
      __builtin_amdgcn_s_wait_tensorcnt(0);
    }
    if (t + 1 < 1024) __builtin_prefetch(px + ((t + 1) << 11), 0, 1);
    __syncthreads();

    v8f acc = {};
    // input projection (K=64): x fragments streamed from L2
    const _Float16* xf = px + (t << 11);
#pragma unroll
    for (int kt = 0; kt < 2; ++kt) {
      v16h a = *(const v16h*)(xf + (((mi << 1) + kt) << 9) + (lane << 4));
      v16h b = *(const v16h*)(sWk + (((gi << 1) + kt) << 9) + (lane << 4));
      acc = __builtin_amdgcn_wmma_f32_16x16x32_f16(false, a, false, b, (short)0,
                                                   acc, false, false);
    }
    // recurrent projection (K=1024): weights resident in LDS
#pragma unroll 4
    for (int kt = 0; kt < 32; ++kt) {
      v16h a = *(const v16h*)(sA + (((mi << 5) + kt) << 9) + (lane << 4));
      v16h b = *(const v16h*)(sW + (((gi << 5) + kt) << 9) + (lane << 4));
      acc = __builtin_amdgcn_wmma_f32_16x16x32_f16(false, a, false, b, (short)0,
                                                   acc, false, false);
    }
    // spill z tile: C layout -> row = r + 8*(lane>=16), col = lane&15
#pragma unroll
    for (int r = 0; r < 8; ++r) {
      int row = r + ((lane >> 4) << 3);
      sZ[(wave << 8) + (row << 4) + (lane & 15)] = acc[r];
    }
    __syncthreads();

    // gate combine: 512 (m,n) cells, 2 per thread
    _Float16* hdst = hbuf + (((t + 1) & 1) * 32768);
    _Float16* sdst = seq + (t << 15);
#pragma unroll
    for (int e0 = 0; e0 < 2; ++e0) {
      int e = tid + (e0 << 8);
      int m = e >> 4, n = e & 15;
      int mh = m >> 4, mr = m & 15;
      int j0 = (wg << 4) + n;
      float zi = sZ[((0 * 2 + mh) << 8) + (mr << 4) + n] + b1[j0];
      float zf = sZ[((1 * 2 + mh) << 8) + (mr << 4) + n] + b1[1024 + j0];
      float zg = sZ[((2 * 2 + mh) << 8) + (mr << 4) + n] + b1[2048 + j0];
      float zo = sZ[((3 * 2 + mh) << 8) + (mr << 4) + n] + b1[3072 + j0];
      float ig = sigf(zi), fg = sigf(zf), gg = sigf(zg), og = sigf(zo);
      float c = fg * sC[e] + ig * gg;
      sC[e] = c;
      float h = og * sigf(c);
      int idx = a_frag_idx(m, j0, 32);
      _Float16 hh = (_Float16)h;
      hdst[idx] = hh;
      sdst[idx] = hh;
    }
    grid_barrier(bar, t, NWG1);
  }
}

// ---------------------------------------------------------------------------
// LSTM2 persistent scan.  32 WGs x 256 threads.  WG owns gate-cols [wg*16,+16)
// of 4 gates over H2=512.  LDS: W2k shard (128K) + W2r shard (64K) + seq
// A-frags (64K) + h2 A-frags (32K) + z (8K) + c (2K) = 298 KB.
// ---------------------------------------------------------------------------
__global__ __launch_bounds__(256) void lstm2_kernel(
    const _Float16* __restrict__ seq,  // [1024][32768]
    const _Float16* __restrict__ pWk,  // [128][32][512]
    const _Float16* __restrict__ pWr,  // [128][16][512]
    const float*    __restrict__ b2,   // [2048]
    _Float16*       __restrict__ hbuf, // [2][16384]
    float*          __restrict__ h2f,  // [32][512]
    int*            __restrict__ bar) {
  extern __shared__ __align__(16) char smem[];
  _Float16* sWk = (_Float16*)smem;              // 4*32*512
  _Float16* sWr = sWk + 4 * 32 * 512;           // 4*16*512
  _Float16* sX  = sWr + 4 * 16 * 512;           // 2*32*512 (seq_t frags)
  _Float16* sH  = sX  + 2 * 32 * 512;           // 2*16*512 (h2 frags)
  float*    sZ  = (float*)(sH + 2 * 16 * 512);  // 8*256
  float*    sC  = sZ + 8 * 256;                 // 512

  const int tid  = threadIdx.x;
  const int wg   = blockIdx.x;
  const int wave = tid >> 5;
  const int lane = tid & 31;
  const int gi   = wave >> 1;
  const int mi   = wave & 1;

  for (int g = 0; g < 4; ++g) {
    const char* s = (const char*)(pWk + (((g * 32 + wg) * 32) << 9));
    char* d = (char*)(sWk + ((g * 32) << 9));
    for (int i = tid * 16; i < 32 * 512 * 2; i += 256 * 16)
      async_ld_b128(d + i, s + i);
  }
  for (int g = 0; g < 4; ++g) {
    const char* s = (const char*)(pWr + (((g * 32 + wg) * 16) << 9));
    char* d = (char*)(sWr + ((g * 16) << 9));
    for (int i = tid * 16; i < 16 * 512 * 2; i += 256 * 16)
      async_ld_b128(d + i, s + i);
  }
  sC[tid] = 0.0f;
  sC[tid + 256] = 0.0f;
  s_wait_async0();
  __syncthreads();

  for (int t = 0; t < 1024; ++t) {
    // TDM broadcasts: wave0 pulls seq_t (64 KB), wave1 pulls h2 frags (32 KB)
    if (wave == 0) {
      tdm_load_1d(sX, seq + (t << 15), 65536);
      __builtin_amdgcn_s_wait_tensorcnt(0);
    } else if (wave == 1) {
      tdm_load_1d(sH, hbuf + ((t & 1) * 16384), 32768);
      __builtin_amdgcn_s_wait_tensorcnt(0);
    }
    __syncthreads();

    v8f acc = {};
    // input projection (K=1024)
#pragma unroll 4
    for (int kt = 0; kt < 32; ++kt) {
      v16h a = *(const v16h*)(sX + (((mi << 5) + kt) << 9) + (lane << 4));
      v16h b = *(const v16h*)(sWk + (((gi << 5) + kt) << 9) + (lane << 4));
      acc = __builtin_amdgcn_wmma_f32_16x16x32_f16(false, a, false, b, (short)0,
                                                   acc, false, false);
    }
    // recurrent projection (K=512)
#pragma unroll 4
    for (int kt = 0; kt < 16; ++kt) {
      v16h a = *(const v16h*)(sH + (((mi << 4) + kt) << 9) + (lane << 4));
      v16h b = *(const v16h*)(sWr + (((gi << 4) + kt) << 9) + (lane << 4));
      acc = __builtin_amdgcn_wmma_f32_16x16x32_f16(false, a, false, b, (short)0,
                                                   acc, false, false);
    }
#pragma unroll
    for (int r = 0; r < 8; ++r) {
      int row = r + ((lane >> 4) << 3);
      sZ[(wave << 8) + (row << 4) + (lane & 15)] = acc[r];
    }
    __syncthreads();

    _Float16* hdst = hbuf + (((t + 1) & 1) * 16384);
#pragma unroll
    for (int e0 = 0; e0 < 2; ++e0) {
      int e = tid + (e0 << 8);
      int m = e >> 4, n = e & 15;
      int mh = m >> 4, mr = m & 15;
      int j0 = (wg << 4) + n;
      float zi = sZ[((0 * 2 + mh) << 8) + (mr << 4) + n] + b2[j0];
      float zf = sZ[((1 * 2 + mh) << 8) + (mr << 4) + n] + b2[512 + j0];
      float zg = sZ[((2 * 2 + mh) << 8) + (mr << 4) + n] + b2[1024 + j0];
      float zo = sZ[((3 * 2 + mh) << 8) + (mr << 4) + n] + b2[1536 + j0];
      float ig = sigf(zi), fg = sigf(zf), gg = sigf(zg), og = sigf(zo);
      float c = fg * sC[e] + ig * gg;
      sC[e] = c;
      float h = og * sigf(c);
      hdst[a_frag_idx(m, j0, 16)] = (_Float16)h;
      if (t == 1023) h2f[m * 512 + j0] = h;
    }
    grid_barrier(bar, t, NWG2);
  }
}

// ---------------------------------------------------------------------------
// FC head: value = sigmoid(h2 @ Wfc + bfc) [32,5]; v = value @ Wv + bv.
// Output flat: v[32], mu[2]=value[0,0:2], cov[4]=value[0,1:5].
// ---------------------------------------------------------------------------
__global__ __launch_bounds__(256) void head_kernel(
    const float* __restrict__ h2, const float* __restrict__ Wfc,
    const float* __restrict__ bfc, const float* __restrict__ Wv,
    const float* __restrict__ bv, float* __restrict__ out) {
  __shared__ float sval[160];
  int tid = threadIdx.x;
  if (tid < 160) {
    int b = tid / 5, jj = tid % 5;
    float s = bfc[jj];
    for (int k = 0; k < 512; ++k) s += h2[b * 512 + k] * Wfc[k * 5 + jj];
    sval[tid] = sigf(s);
  }
  __syncthreads();
  if (tid < 32) {
    float v = bv[0];
#pragma unroll
    for (int jj = 0; jj < 5; ++jj) v += sval[tid * 5 + jj] * Wv[jj];
    out[tid] = v;
  }
  if (tid == 0) {
    out[32] = sval[0];
    out[33] = sval[1];
    out[34] = sval[1];
    out[35] = sval[2];
    out[36] = sval[3];
    out[37] = sval[4];
  }
}

// ---------------------------------------------------------------------------
extern "C" void kernel_launch(void* const* d_in, const int* in_sizes, int n_in,
                              void* d_out, int out_size, void* d_ws,
                              size_t ws_size, hipStream_t stream) {
  const float* state = (const float*)d_in[0];
  const float* W1k   = (const float*)d_in[1];
  const float* W1r   = (const float*)d_in[2];
  const float* b1    = (const float*)d_in[3];
  const float* W2k   = (const float*)d_in[4];
  const float* W2r   = (const float*)d_in[5];
  const float* b2    = (const float*)d_in[6];
  const float* Wfc   = (const float*)d_in[7];
  const float* bfc   = (const float*)d_in[8];
  const float* Wv    = (const float*)d_in[9];
  const float* bv    = (const float*)d_in[10];

  char* ws = (char*)d_ws;
  int*       bar1  = (int*)(ws + OFF_BAR1);
  int*       bar2  = (int*)(ws + OFF_BAR2);
  _Float16*  hb1   = (_Float16*)(ws + OFF_HB1);
  _Float16*  hb2   = (_Float16*)(ws + OFF_HB2);
  float*     h2f   = (float*)(ws + OFF_H2F);
  _Float16*  px    = (_Float16*)(ws + OFF_PX);
  _Float16*  pW1k  = (_Float16*)(ws + OFF_PW1K);
  _Float16*  pW1r  = (_Float16*)(ws + OFF_PW1R);
  _Float16*  pW2k  = (_Float16*)(ws + OFF_PW2K);
  _Float16*  pW2r  = (_Float16*)(ws + OFF_PW2R);
  _Float16*  seq   = (_Float16*)(ws + OFF_SEQ);

  // 1) zero barriers + h ping-pong buffers
  zero_ws_kernel<<<(ZERO_WORDS + 255) / 256, 256, 0, stream>>>((unsigned int*)ws);

  // 2) pack weights / input into f16 WMMA fragments
  pack_b_kernel<<<(64 * 4096 + 255) / 256, 256, 0, stream>>>(W1k, pW1k, 64, 4096);
  pack_b_kernel<<<(1024 * 4096 + 255) / 256, 256, 0, stream>>>(W1r, pW1r, 1024, 4096);
  pack_b_kernel<<<(1024 * 2048 + 255) / 256, 256, 0, stream>>>(W2k, pW2k, 1024, 2048);
  pack_b_kernel<<<(512 * 2048 + 255) / 256, 256, 0, stream>>>(W2r, pW2r, 512, 2048);
  pack_x_kernel<<<(32 * 1024 * 64 + 255) / 256, 256, 0, stream>>>(state, px);

  // 3) LSTM1 persistent scan (64 WGs, 210 KB dynamic LDS)
  size_t lds1 = (4 * 32 * 512 + 4 * 2 * 512 + 2 * 32 * 512) * sizeof(_Float16) +
                (8 * 256 + 512) * sizeof(float);
  lstm1_kernel<<<NWG1, 256, lds1, stream>>>(px, pW1k, pW1r, b1, hb1, seq, bar1);

  // 4) LSTM2 persistent scan (32 WGs, 298 KB dynamic LDS)
  size_t lds2 = (4 * 32 * 512 + 4 * 16 * 512 + 2 * 32 * 512 + 2 * 16 * 512) *
                    sizeof(_Float16) +
                (8 * 256 + 512) * sizeof(float);
  lstm2_kernel<<<NWG2, 256, lds2, stream>>>(seq, pW2k, pW2r, b2, hb2, h2f, bar2);

  // 5) FC head
  head_kernel<<<1, 256, 0, stream>>>(h2f, Wfc, bfc, Wv, bv, (float*)d_out);
}
